// _MoEBlock_7516192768627
// MI455X (gfx1250) — matrix operand order repaired and checked by reading, not compiled, not run on hardware
//
#include <hip/hip_runtime.h>

typedef __attribute__((ext_vector_type(16))) __bf16 v16bf;
typedef __attribute__((ext_vector_type(8)))  float  v8f;

#define TOKENS 8192   // B*S
#define HDIM   2048
#define NEXP   8
#define MAX_TILES 520 // >= sum ceil(count_e/16) worst case (512 + 7)

static __device__ __forceinline__ __bf16 f2bf(float f) { return (__bf16)f; }

// ---------------- kernel 0: zero counts+cursors (16 ints) ----------------
__global__ void moe_init_kernel(int* counts16) {
    if (threadIdx.x < 16) counts16[threadIdx.x] = 0;
}

// ---------------- kernel 1: router (one wave per token) ----------------
__global__ __launch_bounds__(256) void moe_router_kernel(const float* __restrict__ x,
                                                         const float* __restrict__ wr,
                                                         int* __restrict__ idx,
                                                         int* __restrict__ counts) {
    int token = blockIdx.x * (blockDim.x >> 5) + (threadIdx.x >> 5);
    int lane  = threadIdx.x & 31;
    if (token >= TOKENS) return;
    const float* xr = x + (size_t)token * HDIM;
    float acc[NEXP];
#pragma unroll
    for (int e = 0; e < NEXP; ++e) acc[e] = 0.0f;
    for (int h = lane; h < HDIM; h += 32) {
        float xv = xr[h];
#pragma unroll
        for (int e = 0; e < NEXP; ++e) acc[e] += xv * wr[e * HDIM + h];
    }
#pragma unroll
    for (int off = 16; off > 0; off >>= 1) {
#pragma unroll
        for (int e = 0; e < NEXP; ++e) acc[e] += __shfl_xor(acc[e], off, 32);
    }
    if (lane == 0) {
        int best = 0; float bv = acc[0];
#pragma unroll
        for (int e = 1; e < NEXP; ++e) { if (acc[e] > bv) { bv = acc[e]; best = e; } }
        idx[token] = best;
        atomicAdd(&counts[best], 1);
    }
}

// ---------------- kernel 2: exclusive prefix of counts ----------------
__global__ void moe_prefix_kernel(const int* __restrict__ counts, int* __restrict__ offsets) {
    if (threadIdx.x == 0 && blockIdx.x == 0) {
        int s = 0;
        for (int e = 0; e < NEXP; ++e) { offsets[e] = s; s += counts[e]; }
    }
}

// ---------------- kernel 3: scatter tokens into expert groups ----------------
__global__ __launch_bounds__(256) void moe_scatter_kernel(const int* __restrict__ idx,
                                                          const int* __restrict__ offsets,
                                                          int* __restrict__ cursors,
                                                          int* __restrict__ perm) {
    int t = blockIdx.x * blockDim.x + threadIdx.x;
    if (t >= TOKENS) return;
    int e = idx[t];
    int p = atomicAdd(&cursors[e], 1);
    perm[offsets[e] + p] = t;
}

// ---------------- kernel 4: grouped bf16 WMMA GEMM ----------------
// D[tok, f] = sum_k x[tok,k] * W[e][f,k] for the 16 tokens of this tile.
// block = 128 threads (4 waves); each wave computes a 16-token x 32-feature strip
// (two 16x16 WMMA tiles sharing one A fragment -> 2 independent accumulator chains).
__global__ __launch_bounds__(128) void moe_gemm_kernel(const float* __restrict__ x,
                                                       const float* __restrict__ w,
                                                       const int* __restrict__ perm,
                                                       const int* __restrict__ counts,
                                                       const int* __restrict__ offsets,
                                                       float* __restrict__ out) {
    __shared__ int stok[16];

    // Map blockIdx.x -> (expert e, local token-tile r). Uniform across block.
    int tile = blockIdx.x;
    int e = -1, r = 0, acc = 0;
#pragma unroll
    for (int i = 0; i < NEXP; ++i) {
        int te = (counts[i] + 15) >> 4;
        if (e < 0 && tile < acc + te) { e = i; r = tile - acc; }
        acc += te;
    }
    if (e < 0) return; // slot beyond total tiles (uniform exit, before any barrier)

    int cnt = counts[e], off = offsets[e];
    if (threadIdx.x < 16) {
        int g = r * 16 + threadIdx.x;
        stok[threadIdx.x] = (g < cnt) ? perm[off + g] : -1;
    }
    __syncthreads();

    const int wave = threadIdx.x >> 5;
    const int lane = threadIdx.x & 31;
    const int hi   = lane >> 4;      // half-wave select
    const int n    = lane & 15;      // tile row/col within half
    const int pair = blockIdx.y * 4 + wave;  // 0..63 strips of 32 features
    const int fbase0 = pair * 32;
    const int fbase1 = fbase0 + 16;

    // Hoist the token IDs each lane needs (A source + store rows) into registers.
    int tokA = stok[n];
    int tokS[8];
#pragma unroll
    for (int rr = 0; rr < 8; ++rr) tokS[rr] = stok[rr + hi * 8];

    int rowA = tokA < 0 ? 0 : tokA;  // safe address for padded rows (row never stored)
    const float* ap  = x + (size_t)rowA * HDIM;
    const float* bp0 = w + (size_t)e * HDIM * HDIM + (size_t)(fbase0 + n) * HDIM;
    const float* bp1 = w + (size_t)e * HDIM * HDIM + (size_t)(fbase1 + n) * HDIM;

    v8f c0 = {};
    v8f c1 = {};
    for (int kb = 0; kb < HDIM; kb += 32) {
        // ---- A: lanes<16 hold K kb+0..15, lanes>=16 hold K kb+16..31 (matches B)
        int ka0 = kb + hi * 16;
        int ka1 = ka0 + 8;
        float4 a0 = *(const float4*)(ap + ka0);
        float4 a1 = *(const float4*)(ap + ka0 + 4);
        float4 a2 = *(const float4*)(ap + ka1);
        float4 a3 = *(const float4*)(ap + ka1 + 4);
        // ---- B: 32x16 layout: lanes<16 hold K kb+0..15, lanes>=16 hold K kb+16..31 (col = f)
        int kbv = kb + hi * 16;
        float4 p0 = *(const float4*)(bp0 + kbv);
        float4 p1 = *(const float4*)(bp0 + kbv + 4);
        float4 p2 = *(const float4*)(bp0 + kbv + 8);
        float4 p3 = *(const float4*)(bp0 + kbv + 12);
        float4 q0 = *(const float4*)(bp1 + kbv);
        float4 q1 = *(const float4*)(bp1 + kbv + 4);
        float4 q2 = *(const float4*)(bp1 + kbv + 8);
        float4 q3 = *(const float4*)(bp1 + kbv + 12);

        v16bf A, B0, B1;
        A[0]  = f2bf(a0.x); A[1]  = f2bf(a0.y); A[2]  = f2bf(a0.z); A[3]  = f2bf(a0.w);
        A[4]  = f2bf(a1.x); A[5]  = f2bf(a1.y); A[6]  = f2bf(a1.z); A[7]  = f2bf(a1.w);
        A[8]  = f2bf(a2.x); A[9]  = f2bf(a2.y); A[10] = f2bf(a2.z); A[11] = f2bf(a2.w);
        A[12] = f2bf(a3.x); A[13] = f2bf(a3.y); A[14] = f2bf(a3.z); A[15] = f2bf(a3.w);
        B0[0]  = f2bf(p0.x); B0[1]  = f2bf(p0.y); B0[2]  = f2bf(p0.z); B0[3]  = f2bf(p0.w);
        B0[4]  = f2bf(p1.x); B0[5]  = f2bf(p1.y); B0[6]  = f2bf(p1.z); B0[7]  = f2bf(p1.w);
        B0[8]  = f2bf(p2.x); B0[9]  = f2bf(p2.y); B0[10] = f2bf(p2.z); B0[11] = f2bf(p2.w);
        B0[12] = f2bf(p3.x); B0[13] = f2bf(p3.y); B0[14] = f2bf(p3.z); B0[15] = f2bf(p3.w);
        B1[0]  = f2bf(q0.x); B1[1]  = f2bf(q0.y); B1[2]  = f2bf(q0.z); B1[3]  = f2bf(q0.w);
        B1[4]  = f2bf(q1.x); B1[5]  = f2bf(q1.y); B1[6]  = f2bf(q1.z); B1[7]  = f2bf(q1.w);
        B1[8]  = f2bf(q2.x); B1[9]  = f2bf(q2.y); B1[10] = f2bf(q2.z); B1[11] = f2bf(q2.w);
        B1[12] = f2bf(q3.x); B1[13] = f2bf(q3.y); B1[14] = f2bf(q3.z); B1[15] = f2bf(q3.w);

        c0 = __builtin_amdgcn_wmma_f32_16x16x32_bf16(
                 false, A, false, B0, (short)0, c0, false, false);
        c1 = __builtin_amdgcn_wmma_f32_16x16x32_bf16(
                 false, A, false, B1, (short)0, c1, false, false);
    }

    // C/D layout: VGPR rr -> M = rr (lanes 0-15) or rr+8 (lanes 16-31); N = lane%16
#pragma unroll
    for (int rr = 0; rr < 8; ++rr) {
        int tok = tokS[rr];
        if (tok >= 0) {
            float* orow = out + (size_t)tok * HDIM;
            orow[fbase0 + n] = c0[rr];
            orow[fbase1 + n] = c1[rr];
        }
    }
}

extern "C" void kernel_launch(void* const* d_in, const int* in_sizes, int n_in,
                              void* d_out, int out_size, void* d_ws, size_t ws_size,
                              hipStream_t stream) {
    (void)in_sizes; (void)n_in; (void)out_size; (void)ws_size;
    const float* x  = (const float*)d_in[0];   // [B,S,H]
    const float* wr = (const float*)d_in[1];   // [E,H]
    const float* we = (const float*)d_in[2];   // [E,H,H]
    float* out = (float*)d_out;                // [B,S,H]

    int* idx     = (int*)d_ws;          // 8192
    int* perm    = idx + TOKENS;        // 8192
    int* counts  = perm + TOKENS;       // 8
    int* cursors = counts + NEXP;       // 8  (contiguous with counts for init)
    int* offsets = cursors + NEXP;      // 8

    moe_init_kernel<<<1, 32, 0, stream>>>(counts);
    moe_router_kernel<<<TOKENS / 8, 256, 0, stream>>>(x, wr, idx, counts);
    moe_prefix_kernel<<<1, 32, 0, stream>>>(counts, offsets);
    moe_scatter_kernel<<<TOKENS / 256, 256, 0, stream>>>(idx, offsets, cursors, perm);

    dim3 grid(MAX_TILES, 16);
    moe_gemm_kernel<<<grid, 128, 0, stream>>>(x, we, perm, counts, offsets, out);
}
